// CRFusion_52956946759767
// MI455X (gfx1250) — matrix operand order repaired and checked by reading, not compiled, not run on hardware
//
#include <hip/hip_runtime.h>
#include <hip/hip_bf16.h>
#include <math.h>

// ---------------------------------------------------------------------------
// RSFM (6x fused mamba resblocks + in/out 1x1 convs) for MI455X / gfx1250.
// Memory-bound workload: fusing the 6 resblocks keeps the h state
// [B,L,128,16] in registers (one wave = two L positions, 64 h elems/lane),
// cutting ~900MB of h traffic down to a single 75MB final store.
// All GEMMs use v_wmma_f32_16x16x32_bf16 (wave32 WMMA, f32 accumulate).
// Round-2 changes: transpose-on-store X tile (kills ds_load_u16 scalarized
// B fragments), double-buffered K-loop in kernel 1, and a contiguous
// per-lane h mapping in phase G (no cross-lane shuffles, vector LDS reads).
// ---------------------------------------------------------------------------

#define BB   4
#define CIN  1024
#define HWD  48
#define LL   2304            // 48*48
#define DM   64
#define DI   128
#define DTR  32
#define DST  16
#define NTILES (LL / 16)     // 144 tiles of 16 positions per batch
#define PLOC 28              // 16 + 2*6 halo positions per tile
#define PR   32              // padded row count for 2 M-tiles

typedef __bf16 bf16;
typedef __attribute__((ext_vector_type(16))) __bf16 bf16x16;
typedef __attribute__((ext_vector_type(4)))  __bf16 bf16x4;
typedef __attribute__((ext_vector_type(8)))  float  f32x8;

// ---------------- WMMA fragment helpers (contiguous k order, same for A and B) ---------

__device__ __forceinline__ f32x8 zero8() {
  f32x8 z;
#pragma unroll
  for (int j = 0; j < 8; ++j) z[j] = 0.f;
  return z;
}

// A: 16x32 bf16. lane&15 = row M. lanes 0-15 hold K 0..15,
// lanes 16-31 hold K 16..31.  base -> element [row0][k0].
// 32B contiguous per lane, 16B-aligned with our strides -> 2x ds_load_b128.
__device__ __forceinline__ bf16x16 frag_a(const bf16* base, int stride, int lane) {
  const int m = lane & 15, hi = lane >> 4;
  const bf16* r = base + m * stride + hi * 16;
  bf16x16 a;
#pragma unroll
  for (int j = 0; j < 16; ++j) a[j] = r[j];
  return a;
}

// B: 32x16 bf16 stored n-major in LDS as [n][k] rows.
// lane&15 = column N; lanes 0-15 hold K 0..15, lanes 16-31 hold K 16..31.
// 32B contiguous per lane, 16B aligned -> 2x ds_load_b128.
__device__ __forceinline__ bf16x16 frag_b_nmajor(const bf16* base, int stride, int lane) {
  const int n = lane & 15, hi = lane >> 4;
  const bf16* r = base + n * stride + hi * 16;
  bf16x16 b;
#pragma unroll
  for (int j = 0; j < 16; ++j) b[j] = r[j];
  return b;
}

__device__ __forceinline__ f32x8 wmma_bf(bf16x16 a, bf16x16 b, f32x8 c) {
  return __builtin_amdgcn_wmma_f32_16x16x32_bf16(false, a, false, b, (short)0, c,
                                                 false, false);
}

// D/C: 16x16 f32. lane&15 = column N; VGPR r holds row r (lanes<16) / r+8.
__device__ __forceinline__ void frag_store(float* base, int stride, int lane, f32x8 c) {
  const int n = lane & 15, hi = lane >> 4;
#pragma unroll
  for (int r = 0; r < 8; ++r) base[(r + 8 * hi) * stride + n] = c[r];
}

__device__ __forceinline__ float siluf(float x) { return x / (1.f + __expf(-x)); }

// ===========================================================================
// Kernel 1: input 1x1 convs.  rgbp/dtep[b][l][o] = X[b][:,l] . W[o,:] + bias
// GEMM: M = o (4 tiles), N = l (2 tiles of 16), K = c (1024, 32 steps).
// X tile is transposed during the LDS store so B fragments are n-major.
// ===========================================================================

#define K1_WSTRIDE 1032                       // padded halves per W row
#define K1_XSTRIDE 40                         // padded halves per X-tile row
#define K1_LDS ((size_t)(64 * K1_WSTRIDE * 2 + 2 * 32 * K1_XSTRIDE * 2))

__global__ __launch_bounds__(256, 1) void proj_in_kernel(
    const float* __restrict__ rgb, const float* __restrict__ dte,
    const float* __restrict__ w1, const float* __restrict__ b1,
    const float* __restrict__ w2, const float* __restrict__ b2,
    float* __restrict__ rgbp, float* __restrict__ dtep) {
  extern __shared__ char smem[];
  bf16* Wl = (bf16*)smem;                              // [64][1032]
  bf16* Xt = (bf16*)(smem + 64 * K1_WSTRIDE * 2);      // 2 x [32 l][40 c-pad]

  const int bid = blockIdx.x;
  const int inp = bid / (BB * 72);
  const int rem = bid % (BB * 72);
  const int b   = rem / 72;
  const int lc  = rem % 72;                            // chunk of 32 l positions
  const float* X    = inp ? dte : rgb;
  const float* W    = inp ? w2 : w1;
  const float* bias = inp ? b2 : b1;
  float* outp       = inp ? dtep : rgbp;

  const int tid = threadIdx.x, lane = tid & 31, wid = tid >> 5;

  // Cache full weight [64][1024] as bf16 in LDS (padded rows).
  const float4* W4 = (const float4*)W;
  for (int i = tid; i < (64 * 1024) / 4; i += 256) {
    float4 v = W4[i];
    int row = i >> 8;                 // i*4 / 1024
    int col = (i & 255) * 4;
    bf16* d = Wl + row * K1_WSTRIDE + col;
    d[0] = (bf16)v.x; d[1] = (bf16)v.y; d[2] = (bf16)v.z; d[3] = (bf16)v.w;
  }
  __syncthreads();

  const int mt = wid & 3;             // o tile 0..3
  const int nt = wid >> 2;            // l tile 0..1
  const int l0 = lc * 32;
  f32x8 acc = zero8();

  for (int ks = 0; ks < 32; ++ks) {
    bf16* Xcur = Xt + (ks & 1) * (32 * K1_XSTRIDE);
    {
      // 4 coalesced b32 loads (one l per lane, 4 consecutive c per wave),
      // transpose into [l][c] so the B fragment is contiguous n-major.
      const float* src = X + ((size_t)(b * CIN + ks * 32 + wid * 4)) * LL + l0 + lane;
      const float v0 = src[0];
      const float v1 = src[LL];
      const float v2 = src[2 * LL];
      const float v3 = src[3 * LL];
      if (ks < 31) __builtin_prefetch(src + 32 * LL, 0, 1);   // next K slab
      bf16x4 pk;
      pk[0] = (bf16)v0; pk[1] = (bf16)v1; pk[2] = (bf16)v2; pk[3] = (bf16)v3;
      *(bf16x4*)(Xcur + lane * K1_XSTRIDE + wid * 4) = pk;    // one ds_store_b64
    }
    __syncthreads();
    bf16x16 a = frag_a(Wl + (mt * 16) * K1_WSTRIDE + ks * 32, K1_WSTRIDE, lane);
    bf16x16 bq = frag_b_nmajor(Xcur + (nt * 16) * K1_XSTRIDE, K1_XSTRIDE, lane);
    acc = wmma_bf(a, bq, acc);
  }

  const int n = lane & 15, hi = lane >> 4;
  const int l = l0 + nt * 16 + n;
#pragma unroll
  for (int r = 0; r < 8; ++r) {
    const int o = mt * 16 + r + 8 * hi;
    outp[((size_t)(b * LL + l)) * DM + o] = acc[r] + bias[o];
  }
}

// ===========================================================================
// Kernel 2: fused 6x (rmsnorm -> mamba -> residual), h kept in registers.
// One workgroup = 16 waves = one batch b, one tile of 16 L positions
// (+6 halo each side -> PLOC=28 local positions).
// h mapping: wave wid owns positions {wid, wid+16}; lane ln owns the 64
// contiguous state elements idx = ln*64 + j  (d = ln*4 + j/16, s = j&15),
// so the sum over s is lane-private (no shuffles) and h stores are 256B
// contiguous per lane.
// ===========================================================================

// LDS layout (bytes); padded strides chosen for bank-conflict-free frag reads.
enum : int {
  S_WIN  = 256 * 72 * 2,   // in_proj_w  bf16 [256][72]
  S_WX   = 64 * 136 * 2,   // x_proj_w   bf16 [64][136]
  S_WDT  = 128 * 40 * 2,   // dt_proj_w  bf16 [128][40]
  S_WOUT = 64 * 136 * 2,   // out_proj_w bf16 [64][136]
  S_AN   = 128 * 17 * 4,   // -exp(A_log) f32, 17-stride rows
  S_CW   = 384 * 4,        // conv1d_w
  S_CB   = 128 * 4,        // conv1d_b
  S_DTB  = 128 * 4,        // dt_proj_b
  S_DV   = 128 * 4,        // D
  S_RW   = 64 * 4,         // rms_w
  S_XIN  = PR * 72 * 4,    // block input x            f32 [32][72]
  S_XN32 = PR * 72 * 4,    // rmsnorm(x)               f32 [32][72]
  S_XNB  = PR * 72 * 2,    // rmsnorm(x)               bf16
  S_XZ   = PR * 264 * 4,   // in_proj out [x|z]        f32 [32][264]
  S_X1F  = PR * 136 * 4,   // silu(conv1d(x))          f32 [32][136]
  S_X1B  = PR * 136 * 2,   // same, bf16
  S_DBC  = PR * 72 * 4,    // x_proj out (reused for out_proj out)
  S_DLTB = PR * 40 * 2,    // delta_in bf16 [32][40]
  S_DLTF = PR * 136 * 4,   // softplus(dt_proj)        f32 [32][136]
  S_YSB  = PR * 136 * 2,   // y * silu(z)              bf16
};
#define K2_LDS ((size_t)(S_WIN + S_WX + S_WDT + S_WOUT + S_AN + S_CW + S_CB + \
                         S_DTB + S_DV + S_RW + S_XIN + S_XN32 + S_XNB + S_XZ + \
                         S_X1F + S_X1B + S_DBC + S_DLTB + S_DLTF + S_YSB))

__global__ __launch_bounds__(512, 1) void rsfm_fused6_kernel(
    const float* __restrict__ rgbp, const float* __restrict__ dtep,
    const float* __restrict__ rms_w, const float* __restrict__ in_proj_w,
    const float* __restrict__ conv_w, const float* __restrict__ conv_b,
    const float* __restrict__ x_proj_w, const float* __restrict__ dt_w,
    const float* __restrict__ dt_bias, const float* __restrict__ A_log,
    const float* __restrict__ Dvec, const float* __restrict__ out_w,
    float* __restrict__ o6buf, float* __restrict__ hout) {
  extern __shared__ char smem[];
  char* p = smem;
  bf16*  w_in  = (bf16*)p;  p += S_WIN;
  bf16*  w_x   = (bf16*)p;  p += S_WX;
  bf16*  w_dt  = (bf16*)p;  p += S_WDT;
  bf16*  w_out = (bf16*)p;  p += S_WOUT;
  float* An    = (float*)p; p += S_AN;
  float* cw    = (float*)p; p += S_CW;
  float* cb    = (float*)p; p += S_CB;
  float* dtb   = (float*)p; p += S_DTB;
  float* Dv    = (float*)p; p += S_DV;
  float* rw    = (float*)p; p += S_RW;
  float* xin   = (float*)p; p += S_XIN;
  float* xn32  = (float*)p; p += S_XN32;
  bf16*  xnb   = (bf16*)p;  p += S_XNB;
  float* xz    = (float*)p; p += S_XZ;
  float* x1f   = (float*)p; p += S_X1F;
  bf16*  x1b   = (bf16*)p;  p += S_X1B;
  float* dbc   = (float*)p; p += S_DBC;    // [.. :32]=dt_in [32:48]=B [48:64]=C
  bf16*  dltb  = (bf16*)p;  p += S_DLTB;
  float* dltf  = (float*)p; p += S_DLTF;
  bf16*  ysb   = (bf16*)p;  p += S_YSB;

  const int bid   = blockIdx.x;
  const int b     = bid / NTILES;
  const int tile  = bid % NTILES;
  const int lbase = tile * 16 - 6;         // local p -> global l = lbase + p
  const int tid = threadIdx.x, lane = tid & 31, wid = tid >> 5;

  // ---- preload weights / constants into LDS ------------------------------
  for (int i = tid; i < 256 * 64; i += 512) w_in[(i >> 6) * 72 + (i & 63)]  = (bf16)in_proj_w[i];
  for (int i = tid; i < 64 * 128; i += 512) w_x[(i >> 7) * 136 + (i & 127)] = (bf16)x_proj_w[i];
  for (int i = tid; i < 128 * 32; i += 512) w_dt[(i >> 5) * 40 + (i & 31)]  = (bf16)dt_w[i];
  for (int i = tid; i < 64 * 128; i += 512) w_out[(i >> 7) * 136 + (i & 127)] = (bf16)out_w[i];
  for (int i = tid; i < 128 * 16; i += 512) An[(i >> 4) * 17 + (i & 15)] = -__expf(A_log[i]);
  for (int i = tid; i < 384; i += 512) cw[i] = conv_w[i];
  if (tid < 128) { cb[tid] = conv_b[tid]; dtb[tid] = dt_bias[tid]; Dv[tid] = Dvec[tid]; }
  if (tid < 64) rw[tid] = rms_w[tid];
  // zero padding rows 28..31 of bf16 A-operand buffers (read by M-tile 1)
  for (int i = tid; i < 4 * 136; i += 512) {
    x1b[(PLOC + i / 136) * 136 + (i % 136)] = (bf16)0.f;
    ysb[(PLOC + i / 136) * 136 + (i % 136)] = (bf16)0.f;
  }
  for (int i = tid; i < 4 * 72; i += 512) xnb[(PLOC + i / 72) * 72 + (i % 72)] = (bf16)0.f;
  for (int i = tid; i < 4 * 40; i += 512) dltb[(PLOC + i / 40) * 40 + (i % 40)] = (bf16)0.f;

  // ---- initial block input = rgb projection ------------------------------
  for (int i = tid; i < PLOC * DM; i += 512) {
    const int pp = i >> 6, c = i & 63, g = lbase + pp;
    xin[pp * 72 + c] = (g >= 0 && g < LL)
                           ? rgbp[((size_t)(b * LL + g)) * DM + c] : 0.f;
  }

  // ---- h state registers: lane owns idx = lane*64 + j (d=lane*4+j/16) ----
  float h[2][64];
#pragma unroll
  for (int q = 0; q < 2; ++q)
#pragma unroll
    for (int j = 0; j < 64; ++j) h[q][j] = 0.f;

  __syncthreads();

  for (int blk = 0; blk < 6; ++blk) {
    // -- Phase A: rmsnorm (wave per row, 64 channels over 32 lanes x2) -----
    if (wid < 14) {
#pragma unroll
      for (int rr = 0; rr < 2; ++rr) {
        const int pp = wid + rr * 14;     // covers 0..27
        float x0 = xin[pp * 72 + lane];
        float x1v = xin[pp * 72 + 32 + lane];
        float ss = x0 * x0 + x1v * x1v;
        ss += __shfl_xor(ss, 1);  ss += __shfl_xor(ss, 2);
        ss += __shfl_xor(ss, 4);  ss += __shfl_xor(ss, 8);
        ss += __shfl_xor(ss, 16);
        const float sc = rsqrtf(ss * (1.f / 64.f) + 1e-5f);
        const float a0 = x0 * sc * rw[lane];
        const float a1 = x1v * sc * rw[32 + lane];
        xn32[pp * 72 + lane] = a0;       xn32[pp * 72 + 32 + lane] = a1;
        xnb[pp * 72 + lane] = (bf16)a0;  xnb[pp * 72 + 32 + lane] = (bf16)a1;
      }
    }
    __syncthreads();

    // -- Phase B: xz = xn @ in_proj_w^T   (M=2, N=16, K=64) ----------------
    {
      const int nt = wid;                 // 16 N-tiles over 16 waves
#pragma unroll
      for (int mt = 0; mt < 2; ++mt) {
        f32x8 c = zero8();
#pragma unroll
        for (int ks = 0; ks < 2; ++ks) {
          bf16x16 a = frag_a(xnb + mt * 16 * 72 + ks * 32, 72, lane);
          bf16x16 bq = frag_b_nmajor(w_in + nt * 16 * 72 + ks * 32, 72, lane);
          c = wmma_bf(a, bq, c);
        }
        frag_store(xz + mt * 16 * 264 + nt * 16, 264, lane, c);
      }
    }
    __syncthreads();

    // -- Phase C: depthwise conv1d(k=3, pad 1) + silu ----------------------
    for (int i = tid; i < PLOC * DI; i += 512) {
      const int pp = i >> 7, d = i & 127;
      float acc = cb[d];
#pragma unroll
      for (int j = 0; j < 3; ++j) {
        const int q = pp - 1 + j, g = lbase + q;
        const float v = (q >= 0 && q < PLOC && g >= 0 && g < LL)
                            ? xz[q * 264 + d] : 0.f;
        acc += cw[d * 3 + j] * v;
      }
      const float s = siluf(acc);
      x1f[pp * 136 + d] = s;
      x1b[pp * 136 + d] = (bf16)s;
    }
    __syncthreads();

    // -- Phase D: dbc = x1 @ x_proj_w^T   (M=2, N=4, K=128) ----------------
    if (wid < 8) {
      const int mt = wid & 1, nt = wid >> 1;
      f32x8 c = zero8();
#pragma unroll
      for (int ks = 0; ks < 4; ++ks) {
        bf16x16 a = frag_a(x1b + mt * 16 * 136 + ks * 32, 136, lane);
        bf16x16 bq = frag_b_nmajor(w_x + nt * 16 * 136 + ks * 32, 136, lane);
        c = wmma_bf(a, bq, c);
      }
      frag_store(dbc + mt * 16 * 72 + nt * 16, 72, lane, c);
    }
    __syncthreads();

    // -- Phase E: bf16 copy of delta_in ------------------------------------
    for (int i = tid; i < PLOC * DTR; i += 512)
      dltb[(i >> 5) * 40 + (i & 31)] = (bf16)dbc[(i >> 5) * 72 + (i & 31)];
    __syncthreads();

    // -- Phase F: delta = softplus(delta_in @ dt_w^T + b) (M=2,N=8,K=32) ---
    {
      const int mt = wid & 1, nt = wid >> 1;
      f32x8 c = zero8();
      bf16x16 a = frag_a(dltb + mt * 16 * 40, 40, lane);
      bf16x16 bq = frag_b_nmajor(w_dt + nt * 16 * 40, 40, lane);
      c = wmma_bf(a, bq, c);
      const int n = lane & 15, hi = lane >> 4;
      const int dcol = nt * 16 + n;
      const float bia = dtb[dcol];
#pragma unroll
      for (int r = 0; r < 8; ++r) {
        const int m = mt * 16 + r + 8 * hi;
        float v = c[r] + bia;
        v = (v > 20.f) ? v : log1pf(__expf(v));
        dltf[m * 136 + dcol] = v;
      }
    }
    __syncthreads();

    // -- Phase G: state update + y; fully lane-private reduction -----------
    {
#pragma unroll
      for (int q = 0; q < 2; ++q) {
        const int pp = wid + q * 16;
        if (pp < PLOC) {
          // broadcast B/C rows for this position (32 consecutive floats)
          float bmv[16], cmv[16];
#pragma unroll
          for (int s = 0; s < 16; ++s) {
            bmv[s] = dbc[pp * 72 + 32 + s];
            cmv[s] = dbc[pp * 72 + 48 + s];
          }
          // 4 consecutive d-channels per lane -> vector LDS loads
          float dl4[4], xv4[4], z4[4];
#pragma unroll
          for (int dd = 0; dd < 4; ++dd) {
            dl4[dd] = dltf[pp * 136 + lane * 4 + dd];
            xv4[dd] = x1f[pp * 136 + lane * 4 + dd];
            z4[dd]  = xz[pp * 264 + 128 + lane * 4 + dd];
          }
          bf16x4 ys4;
#pragma unroll
          for (int dd = 0; dd < 4; ++dd) {
            const int d = lane * 4 + dd;
            const float dl = dl4[dd];
            const float dlx = dl * xv4[dd];
            float acc = 0.f;
#pragma unroll
            for (int s = 0; s < 16; ++s) {
              const int j = dd * 16 + s;
              const float dA = __expf(dl * An[d * 17 + s]);
              const float hv = dA * h[q][j] + dlx * bmv[s];
              h[q][j] = hv;
              acc += hv * cmv[s];
            }
            acc += Dv[d] * xv4[dd];
            ys4[dd] = (bf16)(acc * siluf(z4[dd]));
          }
          *(bf16x4*)(ysb + pp * 136 + lane * 4) = ys4;   // one ds_store_b64
        }
      }
    }
    __syncthreads();

    // -- Phase H: mo = ys @ out_proj_w^T  (M=2, N=4, K=128), reuse dbc -----
    if (wid < 8) {
      const int mt = wid & 1, nt = wid >> 1;
      f32x8 c = zero8();
#pragma unroll
      for (int ks = 0; ks < 4; ++ks) {
        bf16x16 a = frag_a(ysb + mt * 16 * 136 + ks * 32, 136, lane);
        bf16x16 bq = frag_b_nmajor(w_out + nt * 16 * 136 + ks * 32, 136, lane);
        c = wmma_bf(a, bq, c);
      }
      frag_store(dbc + mt * 16 * 72 + nt * 16, 72, lane, c);
    }
    __syncthreads();

    // -- Phase I: residual + next-block input, or final stores -------------
    if (blk < 5) {
      const float* feat = (blk & 1) ? rgbp : dtep;   // next block's feature
      for (int i = tid; i < PLOC * DM; i += 512) {
        const int pp = i >> 6, c = i & 63, g = lbase + pp;
        const float fv = (g >= 0 && g < LL)
                             ? feat[((size_t)(b * LL + g)) * DM + c] : 0.f;
        xin[pp * 72 + c] = dbc[pp * 72 + c] + xn32[pp * 72 + c] + fv;
      }
    } else {
      // o6 for the 16 valid positions
      for (int i = tid; i < 16 * DM; i += 512) {
        const int pp = 6 + (i >> 6), c = i & 63;
        const int g = lbase + pp;
        o6buf[((size_t)(b * LL + g)) * DM + c] =
            dbc[pp * 72 + c] + xn32[pp * 72 + c];
      }
      // h: flat (d*16+s) == lane*64 + j -> 256B contiguous per lane
#pragma unroll
      for (int q = 0; q < 2; ++q) {
        const int pp = wid + q * 16;
        if (pp >= 6 && pp < 22) {
          const size_t hb = ((size_t)(b * LL + (lbase + pp))) * (DI * DST);
          float* dst = hout + hb + lane * 64;
#pragma unroll
          for (int j = 0; j < 64; ++j) dst[j] = h[q][j];
        }
      }
    }
    __syncthreads();
  }
}

// ===========================================================================
// Kernel 3: out[b][o][l] = o6[b][l][:] . conv3_w[o][:] + conv3_b[o]
// GEMM: M = o (4 tiles/wg, 16 chunks), N = l (2 tiles/wg), K = 64.
// ===========================================================================

#define K3_LDS ((size_t)(64 * 72 * 2 + 32 * 72 * 2))

__global__ __launch_bounds__(256, 1) void proj_out_kernel(
    const float* __restrict__ o6buf, const float* __restrict__ w3,
    const float* __restrict__ b3, float* __restrict__ out) {
  extern __shared__ char smem[];
  bf16* Wl = (bf16*)smem;                    // [64 o][72]
  bf16* Xl = (bf16*)(smem + 64 * 72 * 2);    // [32 l][72]

  const int bid = blockIdx.x;
  const int b   = bid / (16 * 72);
  const int rem = bid % (16 * 72);
  const int mc  = rem / 72;                  // 64-o chunk
  const int lc  = rem % 72;                  // 32-l chunk
  const int tid = threadIdx.x, lane = tid & 31, wid = tid >> 5;

  for (int i = tid; i < 64 * 64; i += 256)
    Wl[(i >> 6) * 72 + (i & 63)] = (bf16)w3[(mc * 64 + (i >> 6)) * DM + (i & 63)];
  for (int i = tid; i < 32 * 64; i += 256)
    Xl[(i >> 6) * 72 + (i & 63)] =
        (bf16)o6buf[((size_t)(b * LL + lc * 32 + (i >> 6))) * DM + (i & 63)];
  __syncthreads();

  const int mt = wid & 3, nt = wid >> 2;
  f32x8 c = zero8();
#pragma unroll
  for (int ks = 0; ks < 2; ++ks) {
    bf16x16 a = frag_a(Wl + mt * 16 * 72 + ks * 32, 72, lane);
    bf16x16 bq = frag_b_nmajor(Xl + nt * 16 * 72 + ks * 32, 72, lane);
    c = wmma_bf(a, bq, c);
  }
  const int n = lane & 15, hi = lane >> 4;
  const int l = lc * 32 + nt * 16 + n;
#pragma unroll
  for (int r = 0; r < 8; ++r) {
    const int o = mc * 64 + mt * 16 + r + 8 * hi;
    out[((size_t)(b * CIN + o)) * LL + l] = c[r] + b3[o];
  }
}

// ===========================================================================
// Host launch
// ===========================================================================

extern "C" void kernel_launch(void* const* d_in, const int* in_sizes, int n_in,
                              void* d_out, int out_size, void* d_ws, size_t ws_size,
                              hipStream_t stream) {
  const float* rgb_feat  = (const float*)d_in[0];
  const float* dte_feat  = (const float*)d_in[1];
  const float* conv1_w   = (const float*)d_in[2];
  const float* conv1_b   = (const float*)d_in[3];
  const float* conv2_w   = (const float*)d_in[4];
  const float* conv2_b   = (const float*)d_in[5];
  const float* conv3_w   = (const float*)d_in[6];
  const float* conv3_b   = (const float*)d_in[7];
  const float* rms_w     = (const float*)d_in[8];
  const float* in_proj_w = (const float*)d_in[9];
  const float* conv1d_w  = (const float*)d_in[10];
  const float* conv1d_b  = (const float*)d_in[11];
  const float* x_proj_w  = (const float*)d_in[12];
  const float* dt_proj_w = (const float*)d_in[13];
  const float* dt_proj_b = (const float*)d_in[14];
  const float* A_log     = (const float*)d_in[15];
  const float* Dvec      = (const float*)d_in[16];
  const float* out_proj_w= (const float*)d_in[17];

  float* ws   = (float*)d_ws;
  float* rgbp = ws;                                // [B][L][64]
  float* dtep = ws + (size_t)BB * LL * DM;         // [B][L][64]
  float* o6   = ws + (size_t)2 * BB * LL * DM;     // [B][L][64]

  float* outp = (float*)d_out;                     // [B][1024][48][48]
  float* hout = outp + (size_t)BB * CIN * LL;      // [B][L][128][16]

  // 1) input projections: 2 inputs x 4 b x 72 l-chunks
  proj_in_kernel<<<2 * BB * 72, 256, K1_LDS, stream>>>(
      rgb_feat, dte_feat, conv1_w, conv1_b, conv2_w, conv2_b, rgbp, dtep);

  // 2) fused 6 resblocks: 4 b x 144 tiles
  rsfm_fused6_kernel<<<BB * NTILES, 512, K2_LDS, stream>>>(
      rgbp, dtep, rms_w, in_proj_w, conv1d_w, conv1d_b, x_proj_w, dt_proj_w,
      dt_proj_b, A_log, Dvec, out_proj_w, o6, hout);

  // 3) output projection: 4 b x 16 o-chunks x 72 l-chunks
  proj_out_kernel<<<BB * 16 * 72, 256, K3_LDS, stream>>>(o6, conv3_w, conv3_b, outp);
}
